// CausalSelfAttention_38817914421555
// MI455X (gfx1250) — matrix operand order repaired and checked by reading, not compiled, not run on hardware
//
#include <hip/hip_runtime.h>

// CDNA5 / gfx1250 causal self-attention, all matmuls on v_wmma_f32_16x16x32_f16.
// Roofline: ~137 GFLOP vs ~150MB HBM traffic -> compute bound -> f16 WMMA path.
// Round 2: 32-row register blocking in GEMM and attention to halve L2-side
// fragment traffic (B/W-fragments and K/V-fragments each reused by two tiles).

typedef __attribute__((ext_vector_type(16))) _Float16 v16h;
typedef __attribute__((ext_vector_type(8)))  _Float16 v8h;
typedef __attribute__((ext_vector_type(4)))  _Float16 v4h;
typedef __attribute__((ext_vector_type(8)))  float    v8f;

#define DEVINL __device__ __forceinline__

constexpr int Bsz = 4;
constexpr int T   = 2048;
constexpr int C   = 1024;
constexpr int H   = 16;
constexpr int Dh  = 64;
constexpr int M   = Bsz * T;          // 8192 token rows
constexpr float SM_SCALE = 0.125f;    // 1/sqrt(64)

DEVINL v16h pack16(v8h lo, v8h hh) {
  v16h a;
#pragma unroll
  for (int i = 0; i < 8; ++i) { a[i] = lo[i]; a[i + 8] = hh[i]; }
  return a;
}

DEVINL v8f wmma_f16(v16h a, v16h b, v8f c) {
  return __builtin_amdgcn_wmma_f32_16x16x32_f16(false, a, false, b, (short)0, c,
                                                false, false);
}

// A-fragment (16x32 f16, M x K): lane row = ln, cols (j&7)+16*(j>>3)+8*hi.
// p = &A[row][kbase]; both loads 16B aligned when kbase % 32 == 0.
DEVINL v16h load_afrag(const _Float16* p, int hi) {
  return pack16(*(const v8h*)(p + 8 * hi), *(const v8h*)(p + 16 + 8 * hi));
}

// B-fragment (32x16 f16, K x N) sourced from a row-major [N][K]-style buffer:
// lane col = ln, rows (j&7)+16*(j>>3)+8*hi. p = &Src[col][kbase].
DEVINL v16h load_bfrag(const _Float16* p, int hi) {
  return pack16(*(const v8h*)(p + 8 * hi), *(const v8h*)(p + 16 + 8 * hi));
}

// ---------------------------------------------------------------- converts ---
__global__ void cvt_f32_f16(const float* __restrict__ in, _Float16* __restrict__ out,
                            int n) {
  int i = (blockIdx.x * 256 + threadIdx.x) * 4;
  if (i >= n) return;
  float4 v = *(const float4*)(in + i);
  v4h h = {(_Float16)v.x, (_Float16)v.y, (_Float16)v.z, (_Float16)v.w};
  *(v4h*)(out + i) = h;
}

// ------------------------------------------------------------------- GEMM ----
// out = A[M x C] (f16, row-major) @ W^T, W row-major [N=C][K=C] (torch Linear).
// One wave computes a 32x64 output block (2x4 D tiles): each B-fragment is
// reused by two WMMA -> 8 WMMA per 12 b128 loads per 32-k step.
// MODE 0: f16 row-major out. MODE 1: f16 transposed per head -> Vt[b][h][d][t].
// MODE 2: f32 out + bias (final projection).
template <int MODE>
__global__ __launch_bounds__(256) void gemm_xwT(const _Float16* __restrict__ A,
                                                const _Float16* __restrict__ W,
                                                void* __restrict__ out,
                                                const float* __restrict__ bias) {
  int w    = (blockIdx.x * 256 + threadIdx.x) >> 5;  // global wave id
  int lane = threadIdx.x & 31;
  int hi = lane >> 4, ln = lane & 15;
  int mb = (w >> 4) * 32;        // 256 M tiles of 32 rows
  int nb = (w & 15) * 64;        // 16 N strips

  v8f acc[2][4] = {};
  const _Float16* arow0 = A + (size_t)(mb + ln) * C;
  const _Float16* arow1 = A + (size_t)(mb + 16 + ln) * C;

  for (int kb = 0; kb < C; kb += 32) {
    v16h a0 = load_afrag(arow0 + kb, hi);
    v16h a1 = load_afrag(arow1 + kb, hi);
#pragma unroll
    for (int t = 0; t < 4; ++t) {
      v16h bf = load_bfrag(W + (size_t)(nb + t * 16 + ln) * C + kb, hi);
      acc[0][t] = wmma_f16(a0, bf, acc[0][t]);
      acc[1][t] = wmma_f16(a1, bf, acc[1][t]);
    }
  }

#pragma unroll
  for (int mi = 0; mi < 2; ++mi) {
    int mrow = mb + mi * 16;
    if (MODE == 0) {
      _Float16* o = (_Float16*)out;
#pragma unroll
      for (int t = 0; t < 4; ++t)
#pragma unroll
        for (int r = 0; r < 8; ++r)
          o[(size_t)(mrow + r + 8 * hi) * C + nb + t * 16 + ln] =
              (_Float16)acc[mi][t][r];
    } else if (MODE == 1) {
      _Float16* o = (_Float16*)out;  // Vt[b][h][d][t]
#pragma unroll
      for (int t = 0; t < 4; ++t) {
        int n = nb + t * 16 + ln;
        int h = n >> 6, d = n & 63;
#pragma unroll
        for (int r = 0; r < 8; ++r) {
          int m = mrow + r + 8 * hi;
          int b = m >> 11, tt = m & 2047;
          o[(((size_t)(b * H + h)) * Dh + d) * T + tt] = (_Float16)acc[mi][t][r];
        }
      }
    } else {
      float* o = (float*)out;
#pragma unroll
      for (int t = 0; t < 4; ++t) {
        float bv = bias[nb + t * 16 + ln];
#pragma unroll
        for (int r = 0; r < 8; ++r)
          o[(size_t)(mrow + r + 8 * hi) * C + nb + t * 16 + ln] =
              acc[mi][t][r] + bv;
      }
    }
  }
}

// -------------------------------------------------------------- attention ----
// One wave per (b, h, 32-query tile) = two 16-row S/O tile sets sharing every
// K- and V-fragment. qb % 32 == 0 -> the final 32-key block is exactly the
// diagonal block for both halves (no fully-masked row). Flash-style online
// softmax; P bounces D-layout -> LDS -> A-layout (same-wave LDS is in-order).
__global__ __launch_bounds__(256) void attn_flash(const _Float16* __restrict__ Q,
                                                  const _Float16* __restrict__ K,
                                                  const _Float16* __restrict__ Vt,
                                                  _Float16* __restrict__ O) {
  __shared__ __align__(16) _Float16 Pst[8][2][16][32];
  int wv   = threadIdx.x >> 5;
  int lane = threadIdx.x & 31;
  int hi = lane >> 4, ln = lane & 15;

  int w   = blockIdx.x * 8 + wv;       // 4096 wave tiles
  int b   = w >> 10;                   // / (H * T/32)
  int rem = w & 1023;
  int h   = rem >> 6;
  int qb  = (rem & 63) * 32;
  int hc  = h * Dh;

  v16h qf[2][2];
#pragma unroll
  for (int u = 0; u < 2; ++u) {
    const _Float16* qrow = Q + (size_t)(b * T + qb + u * 16 + ln) * C + hc;
    qf[u][0] = load_afrag(qrow, hi);       // d = 0..31
    qf[u][1] = load_afrag(qrow + 32, hi);  // d = 32..63
  }

  const _Float16* krows = K + (size_t)(b * T) * C + hc;
  const _Float16* vhead = Vt + ((size_t)(b * H + h)) * Dh * T;

  v8f oacc[2][4] = {};
  float mrow[2][8], lrow[2][8];
#pragma unroll
  for (int u = 0; u < 2; ++u)
#pragma unroll
    for (int r = 0; r < 8; ++r) { mrow[u][r] = -1e30f; lrow[u][r] = 0.0f; }

  int kend = qb + 32;  // causal: keys <= qb+31, always <= T
  for (int kb = 0; kb < kend; kb += 32) {
    // K fragments: B tile (d, t) = K[t][d]; row-major K gives contiguous loads.
    const _Float16* k0 = krows + (size_t)(kb + ln) * C;
    const _Float16* k1 = krows + (size_t)(kb + 16 + ln) * C;
    v16h k0d0 = load_bfrag(k0, hi);
    v16h k0d1 = load_bfrag(k0 + 32, hi);
    v16h k1d0 = load_bfrag(k1, hi);
    v16h k1d1 = load_bfrag(k1 + 32, hi);

    v8f s[2][2] = {};
#pragma unroll
    for (int u = 0; u < 2; ++u) {
      s[u][0] = wmma_f16(qf[u][0], k0d0, s[u][0]);
      s[u][0] = wmma_f16(qf[u][1], k0d1, s[u][0]);
      s[u][1] = wmma_f16(qf[u][0], k1d0, s[u][1]);
      s[u][1] = wmma_f16(qf[u][1], k1d1, s[u][1]);
    }

    float alpha[2][8];
#pragma unroll
    for (int u = 0; u < 2; ++u) {
#pragma unroll
      for (int r = 0; r < 8; ++r) {
        int qi = qb + u * 16 + r + 8 * hi;
        float a0 = (kb + ln <= qi) ? s[u][0][r] * SM_SCALE : -1e30f;
        float a1 = (kb + 16 + ln <= qi) ? s[u][1][r] * SM_SCALE : -1e30f;
        float mx = fmaxf(a0, a1);
#pragma unroll
        for (int off = 8; off; off >>= 1) mx = fmaxf(mx, __shfl_xor(mx, off, 32));
        float mn = fmaxf(mrow[u][r], mx);
        alpha[u][r] = __expf(mrow[u][r] - mn);
        mrow[u][r]  = mn;
        float p0 = __expf(a0 - mn);
        float p1 = __expf(a1 - mn);
        float rs = p0 + p1;
#pragma unroll
        for (int off = 8; off; off >>= 1) rs += __shfl_xor(rs, off, 32);
        lrow[u][r] = lrow[u][r] * alpha[u][r] + rs;
        s[u][0][r] = p0;
        s[u][1][r] = p1;
      }
    }

    // P: D-layout -> LDS -> A-layout (16x32 f16). Same-wave LDS is in-order.
#pragma unroll
    for (int u = 0; u < 2; ++u)
#pragma unroll
      for (int r = 0; r < 8; ++r) {
        Pst[wv][u][r + 8 * hi][ln]      = (_Float16)s[u][0][r];
        Pst[wv][u][r + 8 * hi][16 + ln] = (_Float16)s[u][1][r];
      }
    v16h pa0 = load_afrag(&Pst[wv][0][ln][0], hi);
    v16h pa1 = load_afrag(&Pst[wv][1][ln][0], hi);

#pragma unroll
    for (int t = 0; t < 4; ++t) {
      v16h vf = load_bfrag(vhead + (size_t)(t * 16 + ln) * T + kb, hi);
#pragma unroll
      for (int r = 0; r < 8; ++r) {
        oacc[0][t][r] *= alpha[0][r];
        oacc[1][t][r] *= alpha[1][r];
      }
      oacc[0][t] = wmma_f16(pa0, vf, oacc[0][t]);
      oacc[1][t] = wmma_f16(pa1, vf, oacc[1][t]);
    }
  }

#pragma unroll
  for (int u = 0; u < 2; ++u)
#pragma unroll
    for (int r = 0; r < 8; ++r) {
      float inv = 1.0f / lrow[u][r];
      size_t ob = (size_t)(b * T + qb + u * 16 + r + 8 * hi) * C + hc;
#pragma unroll
      for (int t = 0; t < 4; ++t)
        O[ob + t * 16 + ln] = (_Float16)(oacc[u][t][r] * inv);
    }
}

// ----------------------------------------------------------------- launch ----
extern "C" void kernel_launch(void* const* d_in, const int* in_sizes, int n_in,
                              void* d_out, int out_size, void* d_ws, size_t ws_size,
                              hipStream_t stream) {
  (void)in_sizes; (void)n_in; (void)out_size; (void)ws_size;
  const float* x  = (const float*)d_in[0];
  const float* Wq = (const float*)d_in[1];
  const float* Wk = (const float*)d_in[2];
  const float* Wv = (const float*)d_in[3];
  const float* Wo = (const float*)d_in[4];
  const float* bo = (const float*)d_in[5];

  char* ws = (char*)d_ws;
  _Float16* Xh  = (_Float16*)ws; ws += (size_t)M * C * 2;   // 16 MB
  _Float16* Wqh = (_Float16*)ws; ws += (size_t)C * C * 2;   //  2 MB
  _Float16* Wkh = (_Float16*)ws; ws += (size_t)C * C * 2;
  _Float16* Wvh = (_Float16*)ws; ws += (size_t)C * C * 2;
  _Float16* Woh = (_Float16*)ws; ws += (size_t)C * C * 2;
  _Float16* Qh  = (_Float16*)ws; ws += (size_t)M * C * 2;   // 16 MB
  _Float16* Kh  = (_Float16*)ws; ws += (size_t)M * C * 2;
  _Float16* Vt  = (_Float16*)ws; ws += (size_t)M * C * 2;   // [B][H][D][T]
  _Float16* Ah  = (_Float16*)ws; ws += (size_t)M * C * 2;

  cvt_f32_f16<<<(M * C / 4 + 255) / 256, 256, 0, stream>>>(x, Xh, M * C);
  cvt_f32_f16<<<(C * C / 4 + 255) / 256, 256, 0, stream>>>(Wq, Wqh, C * C);
  cvt_f32_f16<<<(C * C / 4 + 255) / 256, 256, 0, stream>>>(Wk, Wkh, C * C);
  cvt_f32_f16<<<(C * C / 4 + 255) / 256, 256, 0, stream>>>(Wv, Wvh, C * C);
  cvt_f32_f16<<<(C * C / 4 + 255) / 256, 256, 0, stream>>>(Wo, Woh, C * C);

  const int gemm_blocks = (M / 32) * (C / 64) / 8;  // 512 blocks x 8 waves
  gemm_xwT<0><<<gemm_blocks, 256, 0, stream>>>(Xh, Wqh, (void*)Qh, nullptr);
  gemm_xwT<0><<<gemm_blocks, 256, 0, stream>>>(Xh, Wkh, (void*)Kh, nullptr);
  gemm_xwT<1><<<gemm_blocks, 256, 0, stream>>>(Xh, Wvh, (void*)Vt, nullptr);

  const int attn_blocks = (Bsz * H * (T / 32)) / 8;  // 512
  attn_flash<<<attn_blocks, 256, 0, stream>>>(Qh, Kh, Vt, Ah);

  gemm_xwT<2><<<gemm_blocks, 256, 0, stream>>>(Ah, Woh, d_out, bo);
}
